// LSTM_76862734729595
// MI455X (gfx1250) — matrix operand order repaired and checked by reading, not compiled, not run on hardware
//
#include <hip/hip_runtime.h>
#include <hip/hip_bf16.h>

// ---------------- problem constants ----------------
#define BB   256   // batch
#define TT   256   // timesteps
#define IN0  2     // input dim
#define SS   2     // dynamic state dim
#define HH   1024  // hidden
#define ZZ   (SS + HH)      // 1026
#define NG   (4 * HH)       // 4096 gate outputs
#define KW   (IN0 + SS + HH) // 1028 = WU_w row length

// d_out layout (flat, reference return order)
#define OUT_OFF   ((size_t)0)                              // outputs  B*T*Z
#define ZF_OFF    ((size_t)BB * TT * ZZ)                   // zf       B*Z
#define CZF_OFF   (ZF_OFF + (size_t)BB * ZZ)               // czf      B*Z
#define COEFF_OFF (CZF_OFF + (size_t)BB * ZZ)              // coeffs   B*T*S
#define MATS_OFF  (COEFF_OFF + (size_t)BB * TT * SS)       // mats     T*2*2

typedef __attribute__((ext_vector_type(16))) __bf16 v16bf;
typedef __attribute__((ext_vector_type(8)))  float  v8f;

union ABu {
    unsigned int u[8];
    v16bf        bf;
};

__device__ __forceinline__ unsigned short f2bf(float f) {
    unsigned int u = __float_as_uint(f);
    unsigned int r = u + 0x7FFFu + ((u >> 16) & 1u);   // round-to-nearest-even
    return (unsigned short)(r >> 16);
}
__device__ __forceinline__ unsigned int packbf2(float a, float b) {
    return (unsigned int)f2bf(a) | ((unsigned int)f2bf(b) << 16);
}
__device__ __forceinline__ float sigmoidf(float x) {
    return 1.0f / (1.0f + expf(-x));
}

// ---------------------------------------------------------------------------
// Repack Wh = WU_w[:, 4:1028] (4096 x 1024, fp32) into bf16 WMMA-B tile order.
// Tile (ntile 0..255, ktile 0..31) = 256 dwords stored as [lane][v]:
//   n = ntile*16 + (lane&15); k = ktile*32 + ((lane>=16)?8:0) + 2(v&3) + ((v>=4)?16:0)  (packed pair)
// so a wave reads its 8 B dwords as two contiguous b128 loads per lane.
// ---------------------------------------------------------------------------
__global__ __launch_bounds__(256) void k_repack(const float* __restrict__ WU_w,
                                                unsigned int* __restrict__ Whp) {
    int idx   = blockIdx.x * 256 + threadIdx.x;   // 2,097,152 total
    int v     = idx & 7;
    int lane  = (idx >> 3) & 31;
    int ktile = (idx >> 8) & 31;
    int ntile = idx >> 13;
    int n = ntile * 16 + (lane & 15);
    int k = ktile * 32 + ((lane >> 4) << 3) + 2 * (v & 3) + ((v >> 2) << 4);
    const float* row = WU_w + (size_t)n * KW + 4;   // skip the 4 "u" columns
    Whp[idx] = packbf2(row[k], row[k + 1]);
}

// ---------------------------------------------------------------------------
// State init from z0 / c_z0.
// ---------------------------------------------------------------------------
__global__ __launch_bounds__(256) void k_init(const float* __restrict__ z0,
                                              const float* __restrict__ c_z0,
                                              float* __restrict__ xstate,
                                              float* __restrict__ hstate,
                                              unsigned short* __restrict__ hbf,
                                              float* __restrict__ cstate) {
    int idx = blockIdx.x * 256 + threadIdx.x;      // 1026*256 = 262,656
    int b = idx / ZZ, j = idx % ZZ;
    float zv = z0[(size_t)b * ZZ + j];
    if (j < SS) {
        xstate[b * SS + j] = zv;
    } else {
        int k = j - SS;
        hstate[(size_t)b * HH + k] = zv;
        hbf[(size_t)b * HH + k]    = f2bf(zv);
        cstate[(size_t)b * HH + k] = c_z0[(size_t)b * ZZ + j];
    }
}

// ---------------------------------------------------------------------------
// Per-step gate GEMM: gates[B,4096] = h_bf16[B,1024] @ Wh^T (bf16 WMMA, f32 acc)
//                                   + [x_prev, inp_t] @ Wu^T + bias  (epilogue)
// Grid (16, 32), 256 threads (8 waves). blockIdx.x = mtile, wave -> ntile.
// A strip (16 rows x 1024 K, bf16 = 32 KB) staged in LDS, shared by 8 waves.
// ---------------------------------------------------------------------------
__global__ __launch_bounds__(256) void k_gates(const unsigned int* __restrict__ Whp,
                                               const unsigned short* __restrict__ hbf,
                                               const float* __restrict__ WU_w,
                                               const float* __restrict__ WU_b,
                                               const float* __restrict__ xstate,
                                               const float* __restrict__ rnn_input,
                                               int t,
                                               float* __restrict__ gates) {
    __shared__ unsigned int sA[16 * 512];          // 16 rows x 512 dwords = 32 KB

    const int lane  = threadIdx.x & 31;
    const int wave  = threadIdx.x >> 5;
    const int mtile = blockIdx.x;                  // 0..15
    const int ntile = blockIdx.y * 8 + wave;       // 0..255
    const int hi    = lane >> 4;                   // lane half
    const int mrow  = lane & 15;

    // cooperative load of A strip (rows mtile*16 .. +15) as b128s
    {
        const uint4* g  = (const uint4*)hbf;       // row = 128 uint4
        uint4*       s4 = (uint4*)sA;
        for (int i = threadIdx.x; i < 2048; i += 256)
            s4[i] = g[(size_t)(mtile * 16 + (i >> 7)) * 128 + (i & 127)];
    }
    __syncthreads();

    v8f acc = {};
    const uint4* sa = (const uint4*)sA;
    const uint4* pb = (const uint4*)Whp;           // tile = 64 uint4

    #pragma unroll 4
    for (int kt = 0; kt < 32; ++kt) {
        ABu A, Bm;
        // A: 16-bit 16x32 layout. v0..3 -> K = hi*8 + 2v (+ktile*32), v4..7 -> +16
        int a4 = (mrow * 512 + kt * 16 + hi * 4) >> 2;     // uint4 index
        uint4 a0 = sa[a4];
        uint4 a1 = sa[a4 + 2];
        A.u[0] = a0.x; A.u[1] = a0.y; A.u[2] = a0.z; A.u[3] = a0.w;
        A.u[4] = a1.x; A.u[5] = a1.y; A.u[6] = a1.z; A.u[7] = a1.w;
        // B: repacked tile, lane-contiguous 32 bytes
        size_t b4 = (size_t)(ntile * 32 + kt) * 64 + lane * 2;
        uint4 b0 = pb[b4];
        uint4 b1 = pb[b4 + 1];
        Bm.u[0] = b0.x; Bm.u[1] = b0.y; Bm.u[2] = b0.z; Bm.u[3] = b0.w;
        Bm.u[4] = b1.x; Bm.u[5] = b1.y; Bm.u[6] = b1.z; Bm.u[7] = b1.w;

        acc = __builtin_amdgcn_wmma_f32_16x16x32_bf16(
            /*neg_a=*/false, A.bf, /*neg_b=*/false, Bm.bf,
            /*c_mod=*/(short)0, acc, /*reuse_a=*/false, /*reuse_b=*/false);
    }

    // epilogue: bias + K=4 "u" part. C layout: VGPR r -> M = r + hi*8, N = lane&15.
    const int n  = ntile * 16 + (lane & 15);
    const float bias = WU_b[n];
    const float w0 = WU_w[(size_t)n * KW + 0];
    const float w1 = WU_w[(size_t)n * KW + 1];
    const float w2 = WU_w[(size_t)n * KW + 2];
    const float w3 = WU_w[(size_t)n * KW + 3];
    #pragma unroll
    for (int r = 0; r < 8; ++r) {
        int mm = mtile * 16 + r + hi * 8;
        float x0 = xstate[mm * SS + 0];
        float x1 = xstate[mm * SS + 1];
        float i0 = rnn_input[(size_t)mm * TT * IN0 + t * IN0 + 0];
        float i1 = rnn_input[(size_t)mm * TT * IN0 + t * IN0 + 1];
        gates[(size_t)mm * NG + n] = acc[r] + bias + w0 * x0 + w1 * x1 + w2 * i0 + w3 * i1;
    }
}

// ---------------------------------------------------------------------------
// LSTM elementwise cell update; also emits h_new as bf16 for next step's A and
// writes the h part of outputs[:, t, 2:].
// ---------------------------------------------------------------------------
__global__ __launch_bounds__(256) void k_cell(const float* __restrict__ gates,
                                              float* __restrict__ cstate,
                                              float* __restrict__ hstate,
                                              unsigned short* __restrict__ hbf,
                                              float* __restrict__ out,
                                              int t) {
    int idx = blockIdx.x * 256 + threadIdx.x;      // 262,144
    int b = idx >> 10, k = idx & (HH - 1);
    const float* g = gates + (size_t)b * NG;
    float gi = g[k];
    float gf = g[HH + k];
    float gg = g[2 * HH + k];
    float go = g[3 * HH + k];
    float i = sigmoidf(gi);
    float f = sigmoidf(gf);
    float gv = tanhf(gg);
    float o = sigmoidf(go);
    float c = cstate[idx];
    float cn = f * c + i * gv;
    float hn = o * tanhf(cn);
    cstate[idx] = cn;
    hstate[idx] = hn;
    hbf[idx]    = f2bf(hn);
    out[OUT_OFF + (size_t)b * TT * ZZ + (size_t)t * ZZ + SS + k] = hn;
}

// ---------------------------------------------------------------------------
// Projections (alpha, Wx: 4 dots of length 1024 per batch row, LDS reduction),
// 2x2 dynamics, x update; writes outputs x part, coefficients, mats.
// One block per batch element.
// ---------------------------------------------------------------------------
__global__ __launch_bounds__(256) void k_proj(const float* __restrict__ hstate,
                                              const float* __restrict__ alpha_w,
                                              const float* __restrict__ alpha_b,
                                              const float* __restrict__ Wx_w,
                                              const float* __restrict__ Wx_b,
                                              float* __restrict__ xstate,
                                              const float* __restrict__ rnn_input,
                                              const float* __restrict__ tau,
                                              float* __restrict__ out,
                                              int t) {
    __shared__ float red0[256], red1[256], red2[256], red3[256];
    int b = blockIdx.x, tid = threadIdx.x;
    float s0 = 0.f, s1 = 0.f, s2 = 0.f, s3 = 0.f;
    const float* h = hstate + (size_t)b * HH;
    for (int k = tid; k < HH; k += 256) {
        float hv = h[k];
        s0 += hv * alpha_w[k];
        s1 += hv * alpha_w[HH + k];
        s2 += hv * Wx_w[k];
        s3 += hv * Wx_w[HH + k];
    }
    red0[tid] = s0; red1[tid] = s1; red2[tid] = s2; red3[tid] = s3;
    __syncthreads();
    for (int off = 128; off > 0; off >>= 1) {
        if (tid < off) {
            red0[tid] += red0[tid + off];
            red1[tid] += red1[tid + off];
            red2[tid] += red2[tid + off];
            red3[tid] += red3[tid + off];
        }
        __syncthreads();
    }
    if (tid == 0) {
        float a0 = sigmoidf(red0[0] + alpha_b[0]);
        float a1 = sigmoidf(red1[0] + alpha_b[1]);
        float p0 = red2[0] + Wx_b[0];
        float p1 = red3[0] + Wx_b[1];
        float x1 = xstate[b * SS + 0];
        float x2 = xstate[b * SS + 1];
        float u  = rnn_input[(size_t)b * TT * IN0 + t * IN0 + 0];
        float tt = tau[(size_t)b * TT + t];
        float a21 = -1.f - 2.f * x1 * x2;   // MHU = DYN = 1
        float a22 = 1.f - x1 * x1;
        float dx0 = x2;
        float dx1 = a21 * x1 + a22 * x2 + u;
        float xm0 = x1 + tt * dx0;
        float xm1 = x2 + tt * dx1;
        float xn0 = (1.f - a0) * xm0 + a0 * p0;
        float xn1 = (1.f - a1) * xm1 + a1 * p1;
        xstate[b * SS + 0] = xn0;
        xstate[b * SS + 1] = xn1;
        size_t ob = OUT_OFF + (size_t)b * TT * ZZ + (size_t)t * ZZ;
        out[ob + 0] = xn0;
        out[ob + 1] = xn1;
        size_t cb = COEFF_OFF + (size_t)b * TT * SS + (size_t)t * SS;
        out[cb + 0] = a0;
        out[cb + 1] = a1;
        if (b == 0) {
            size_t mb = MATS_OFF + (size_t)t * 4;
            out[mb + 0] = 0.f;
            out[mb + 1] = 1.f;
            out[mb + 2] = a21;
            out[mb + 3] = a22;
        }
    }
}

// ---------------------------------------------------------------------------
// Final states zf / czf. czf[:, :2] never changes (== c_z0[:, :2]).
// ---------------------------------------------------------------------------
__global__ __launch_bounds__(256) void k_final(const float* __restrict__ xstate,
                                               const float* __restrict__ hstate,
                                               const float* __restrict__ cstate,
                                               const float* __restrict__ c_z0,
                                               float* __restrict__ out) {
    int idx = blockIdx.x * 256 + threadIdx.x;      // 262,656
    int b = idx / ZZ, j = idx % ZZ;
    float zv, cv;
    if (j < SS) {
        zv = xstate[b * SS + j];
        cv = c_z0[(size_t)b * ZZ + j];
    } else {
        zv = hstate[(size_t)b * HH + (j - SS)];
        cv = cstate[(size_t)b * HH + (j - SS)];
    }
    out[ZF_OFF  + (size_t)idx] = zv;
    out[CZF_OFF + (size_t)idx] = cv;
}

// ---------------------------------------------------------------------------
extern "C" void kernel_launch(void* const* d_in, const int* in_sizes, int n_in,
                              void* d_out, int out_size, void* d_ws, size_t ws_size,
                              hipStream_t stream) {
    const float* rnn_input = (const float*)d_in[0];
    const float* tau       = (const float*)d_in[1];
    const float* z0        = (const float*)d_in[2];
    const float* c_z0      = (const float*)d_in[3];
    const float* WU_w      = (const float*)d_in[4];
    const float* WU_b      = (const float*)d_in[5];
    const float* alpha_w   = (const float*)d_in[6];
    const float* alpha_b   = (const float*)d_in[7];
    const float* Wx_w      = (const float*)d_in[8];
    const float* Wx_b      = (const float*)d_in[9];
    float* out = (float*)d_out;

    // workspace carving
    char* ws = (char*)d_ws;
    unsigned int*   Whp    = (unsigned int*)ws;                        // 8 MB
    unsigned short* hbf    = (unsigned short*)(ws + (8u << 20));       // 512 KB
    float*          hstate = (float*)(ws + (8u << 20) + (512u << 10)); // 1 MB
    float*          cstate = hstate + (size_t)BB * HH;                 // 1 MB
    float*          gates  = cstate + (size_t)BB * HH;                 // 4 MB
    float*          xstate = gates + (size_t)BB * NG;                  // 2 KB

    k_repack<<<8192, 256, 0, stream>>>(WU_w, Whp);
    k_init<<<(BB * ZZ) / 256, 256, 0, stream>>>(z0, c_z0, xstate, hstate, hbf, cstate);

    for (int t = 0; t < TT; ++t) {
        k_gates<<<dim3(16, 32), 256, 0, stream>>>(Whp, hbf, WU_w, WU_b,
                                                  xstate, rnn_input, t, gates);
        k_cell<<<(BB * HH) / 256, 256, 0, stream>>>(gates, cstate, hstate, hbf, out, t);
        k_proj<<<BB, 256, 0, stream>>>(hstate, alpha_w, alpha_b, Wx_w, Wx_b,
                                       xstate, rnn_input, tau, out, t);
    }

    k_final<<<(BB * ZZ) / 256, 256, 0, stream>>>(xstate, hstate, cstate, c_z0, out);
}
